// CrossAttention_84765474554543
// MI455X (gfx1250) — matrix operand order repaired and checked by reading, not compiled, not run on hardware
//
#include <hip/hip_runtime.h>

// ---------------------------------------------------------------------------
// CDNA5 (gfx1250, wave32) cross-attention:
//   Q = (x@Wq+bq)*scale ; K,V = ctx@Wkv+bkv ; O = softmax(QK^T)V ; out = O@Wproj+bproj
// bf16 WMMA (v_wmma_f32_16x16x32_bf16) everywhere, f32 accumulate.
// Data movement: TDM (tensor_load_to_lds) for attention Q/K tiles,
// global_load_async_to_lds_b128 + double buffering for GEMM tiles,
// with compile-time fallbacks if the builtins are unavailable.
// ---------------------------------------------------------------------------

typedef __attribute__((ext_vector_type(16))) __bf16        v16bf;
typedef __attribute__((ext_vector_type(8)))  float         v8f;
typedef __attribute__((ext_vector_type(8)))  unsigned int  v8u;

static_assert(sizeof(v16bf) == 32, "v16bf must be 8 VGPRs");
static_assert(sizeof(v8f)   == 32, "v8f must be 8 VGPRs");

#if defined(__gfx1250__)
#define HAS_ASYNC_LDS (__has_builtin(__builtin_amdgcn_global_load_async_to_lds_b128) && \
                       __has_builtin(__builtin_amdgcn_s_wait_asynccnt))
#define HAS_TDM (__has_builtin(__builtin_amdgcn_tensor_load_to_lds) && \
                 __has_builtin(__builtin_amdgcn_s_wait_tensorcnt))
#else
#define HAS_ASYNC_LDS 0
#define HAS_TDM 0
#endif

typedef __attribute__((ext_vector_type(4))) int          i32x4;
typedef __attribute__((ext_vector_type(8))) int          i32x8;
typedef __attribute__((ext_vector_type(4))) unsigned int u32x4;

// ---- async global -> LDS copy (16B per lane), with synchronous fallback ----
__device__ __forceinline__ void cp_b128(void* lds, const void* g) {
#if HAS_ASYNC_LDS
  __builtin_amdgcn_global_load_async_to_lds_b128(
      (__attribute__((address_space(1))) i32x4*)g,
      (__attribute__((address_space(3))) i32x4*)lds, 0, 0);
#else
  *(uint4*)lds = *(const uint4*)g;
#endif
}
__device__ __forceinline__ void wait_cp() {
#if HAS_ASYNC_LDS
  __builtin_amdgcn_s_wait_asynccnt(0);
#endif
}

// ---- Tensor Data Mover: 2D tile load, one issue per workgroup ----
#if HAS_TDM
// g1w0 packs: workgroup_mask=0, data_size, pad config.
// For 64x bf16 rows padded to 144B: data_size=1(2B) | pad_enable |
// pad_interval=4 (32 DWORDs) | pad_amount=3 (4 DWORDs)  => 0x07110000
__device__ __forceinline__ void tdm_load_2d(unsigned lds_byte_off, const void* gaddr,
                                            unsigned tensor_d0, unsigned tensor_d1,
                                            unsigned tile_d0, unsigned tile_d1,
                                            unsigned row_stride_elems, unsigned g1w0) {
  unsigned long long ga = (unsigned long long)gaddr;
  u32x4 g0;
  g0[0] = 1u;                                                  // count=1, user mode
  g0[1] = lds_byte_off;                                        // lds_addr
  g0[2] = (unsigned)ga;                                        // global_addr[31:0]
  g0[3] = (unsigned)((ga >> 32) & 0x01FFFFFFu) | 0x80000000u;  // addr hi | type=2
  i32x8 g1;
  g1[0] = (int)g1w0;
  g1[1] = (int)((tensor_d0 & 0xFFFFu) << 16);                  // tensor_dim0[15:0]
  g1[2] = (int)((tensor_d0 >> 16) | ((tensor_d1 & 0xFFFFu) << 16));
  g1[3] = (int)((tensor_d1 >> 16) | (tile_d0 << 16));          // tile_dim0
  g1[4] = (int)(tile_d1 & 0xFFFFu);                            // tile_dim1, tile_dim2=0
  g1[5] = (int)row_stride_elems;                               // tensor_dim0_stride lo
  g1[6] = 0;
  g1[7] = 0;
  i32x4 z4 = {0, 0, 0, 0};
  i32x8 z8 = {0, 0, 0, 0, 0, 0, 0, 0};
  __builtin_amdgcn_tensor_load_to_lds(g0, g1, z4, z4, z8, 0);
}
__device__ __forceinline__ unsigned lds_off(const void* p) {
  // generic VA of LDS = {SHARED_BASE, byte_offset}; low 32 bits are the offset
  return (unsigned)(unsigned long long)p;
}
#endif

__device__ __forceinline__ v8f wmma_bf16(v16bf a, v16bf b, v8f c) {
  return __builtin_amdgcn_wmma_f32_16x16x32_bf16(false, a, false, b,
                                                 (short)0, c, false, false);
}

// A-fragment (16x32 bf16) from LDS [row][k], row stride `ld` (halfs).
// lanes 0-15: row M=lane, K={0..7,16..23}; lanes 16-31: row M=lane-16, K={8..15,24..31}
__device__ __forceinline__ v16bf fragA_lds(const __bf16* base, int ld,
                                           int row0, int kbase, int lane) {
  const int half = lane >> 4, m = lane & 15;
  const __bf16* rp = base + (row0 + m) * ld + kbase;
  v8u t;
#pragma unroll
  for (int p = 0; p < 8; ++p) {
    int k = ((p >> 2) << 4) + half * 8 + ((p & 3) << 1);
    t[p] = *(const unsigned*)(rp + k);
  }
  return __builtin_bit_cast(v16bf, t);
}

// B-fragment (32x16 bf16) from LDS stored transposed: base[(col)*ld + k].
// lanes 0-15: col N=lane, K={0..7,16..23}; lanes 16-31: K={8..15,24..31}
__device__ __forceinline__ v16bf fragB_lds(const __bf16* base, int ld,
                                           int col0, int kbase, int lane) {
  const int half = lane >> 4, n = lane & 15;
  const __bf16* cp = base + (col0 + n) * ld + kbase;
  v8u t;
#pragma unroll
  for (int p = 0; p < 8; ++p)
    t[p] = *(const unsigned*)(cp + ((p >> 2) << 4) + half * 8 + ((p & 3) << 1));
  return __builtin_bit_cast(v16bf, t);
}

// ---------------------------------------------------------------------------
// fp32 -> bf16 elementwise convert
// ---------------------------------------------------------------------------
__global__ void __launch_bounds__(256)
cvt_f32_bf16_v4(const float4* __restrict__ in, ushort4* __restrict__ out, long n4) {
  long i = (long)blockIdx.x * 256 + threadIdx.x;
  if (i < n4) {
    float4 v = in[i];
    ushort4 o;
    o.x = __builtin_bit_cast(unsigned short, (__bf16)v.x);
    o.y = __builtin_bit_cast(unsigned short, (__bf16)v.y);
    o.z = __builtin_bit_cast(unsigned short, (__bf16)v.z);
    o.w = __builtin_bit_cast(unsigned short, (__bf16)v.w);
    out[i] = o;
  }
}

// ---------------------------------------------------------------------------
// Transpose + convert: W f32 [Kd][Nn] -> Wt bf16 [Nn][Kd]
// ---------------------------------------------------------------------------
__global__ void __launch_bounds__(256)
transpose_cvt(const float* __restrict__ W, __bf16* __restrict__ Wt, int Kd, int Nn) {
  __shared__ float tile[32][33];
  const int k0 = blockIdx.y * 32, n0 = blockIdx.x * 32;
  const int t = threadIdx.x;
  const int r = t >> 5, c = t & 31;
#pragma unroll
  for (int i = 0; i < 4; ++i)
    tile[r + 8 * i][c] = W[(size_t)(k0 + r + 8 * i) * Nn + n0 + c];
  __syncthreads();
#pragma unroll
  for (int i = 0; i < 4; ++i) {
    int nn = r + 8 * i;
    Wt[(size_t)(n0 + nn) * Kd + k0 + c] = (__bf16)tile[c][nn];
  }
}

// ---------------------------------------------------------------------------
// GEMM: C[m,n] = (sum_k A[m,k]*Wt[n,k] + bias[n]) * outScale
// Double-buffered LDS, async global->LDS staging, 128x128 tile, BK=32, 8 waves.
// ---------------------------------------------------------------------------
__global__ void __launch_bounds__(256)
gemm_bf16(const __bf16* __restrict__ A, const __bf16* __restrict__ Wt,
          const float* __restrict__ bias, __bf16* __restrict__ outb,
          float* __restrict__ outf, float outScale, int Nout, int Kd) {
  __shared__ __attribute__((aligned(16))) __bf16 lA[2][128 * 40];
  __shared__ __attribute__((aligned(16))) __bf16 lB[2][128 * 40];
  const int m0 = blockIdx.y * 128, n0 = blockIdx.x * 128;
  const int t = threadIdx.x, lane = t & 31, w = t >> 5;
  const int wm = (w & 3) * 32;   // wave row offset in tile
  const int wn = (w >> 2) * 64;  // wave col offset in tile
  const int srow = t >> 2, scg = (t & 3) * 8;  // staging: 2 chunks/thread

  v8f acc[2][4];
#pragma unroll
  for (int i = 0; i < 2; ++i)
#pragma unroll
    for (int j = 0; j < 4; ++j)
      acc[i][j] = (v8f){0.f, 0.f, 0.f, 0.f, 0.f, 0.f, 0.f, 0.f};

  auto stage = [&](int buf, int k0) {
#pragma unroll
    for (int i = 0; i < 2; ++i) {
      int row = srow + i * 64;
      cp_b128(&lA[buf][row * 40 + scg], &A[(size_t)(m0 + row) * Kd + k0 + scg]);
      cp_b128(&lB[buf][row * 40 + scg], &Wt[(size_t)(n0 + row) * Kd + k0 + scg]);
    }
  };

  const int nk = Kd >> 5;
  stage(0, 0);
  for (int kt = 0; kt < nk; ++kt) {
    wait_cp();
    __syncthreads();  // staged data visible; previous reads of other buffer done
    if (kt + 1 < nk) stage((kt + 1) & 1, (kt + 1) * 32);

    const __bf16* bufA = lA[kt & 1];
    const __bf16* bufB = lB[kt & 1];
    v16bf a0 = fragA_lds(bufA, 40, wm, 0, lane);
    v16bf a1 = fragA_lds(bufA, 40, wm + 16, 0, lane);
#pragma unroll
    for (int j = 0; j < 4; ++j) {
      v16bf bj = fragB_lds(bufB, 40, wn + 16 * j, 0, lane);
      acc[0][j] = wmma_bf16(a0, bj, acc[0][j]);
      acc[1][j] = wmma_bf16(a1, bj, acc[1][j]);
    }
  }

  const int half = lane >> 4, n = lane & 15;
#pragma unroll
  for (int i = 0; i < 2; ++i) {
#pragma unroll
    for (int j = 0; j < 4; ++j) {
      int col = n0 + wn + 16 * j + n;
      float bv = bias[col];
#pragma unroll
      for (int r = 0; r < 8; ++r) {
        int row = m0 + wm + 16 * i + r + 8 * half;
        float v = (acc[i][j][r] + bv) * outScale;
        if (outf) outf[(size_t)row * Nout + col] = v;
        else      outb[(size_t)row * Nout + col] = (__bf16)v;
      }
    }
  }
}

// ---------------------------------------------------------------------------
// Flash attention per (b, h, 128 query rows). 8 waves x 16 rows.
// Qb: [B,N,1024] bf16 (pre-scaled), KVb: [B,K,2048] bf16 (K|V), Ob: [B,N,1024].
// Q tile and K tiles staged by the Tensor Data Mover (hardware row padding
// 128B data + 16B pad = 144B = LDS row stride 72 halfs); V staged manually
// (needs transpose scatter).
// ---------------------------------------------------------------------------
__global__ void __launch_bounds__(256)
attn_kernel(const __bf16* __restrict__ Qb, const __bf16* __restrict__ KVb,
            __bf16* __restrict__ Ob, int Nq, int Kk) {
  constexpr int D = 1024;
  __shared__ __attribute__((aligned(16))) __bf16 lQ[128 * 72];    // [qrow][d]
  __shared__ __attribute__((aligned(16))) __bf16 lK[32 * 72];     // [key][d]
  __shared__ __attribute__((aligned(16))) __bf16 lVt[64 * 40];    // [d][key]
  __shared__ __attribute__((aligned(16))) __bf16 lP[8][16 * 36];  // per-wave [qrow][key]

  const int b = blockIdx.z, h = blockIdx.y, nt0 = blockIdx.x * 128;
  const int t = threadIdx.x, lane = t & 31, w = t >> 5;
  const int half = lane >> 4, n = lane & 15;

  const size_t qbase = ((size_t)b * Nq + nt0) * D + h * 64;
  const size_t kvb = (size_t)b * Kk * 2048 + h * 64;

  // ---- stage Q tile 128x64 ----
#if HAS_TDM
  if (w == 0) {
    tdm_load_2d(lds_off(lQ), Qb + qbase, 64, 128, 64, 128, D, 0x07110000u);
    __builtin_amdgcn_s_wait_tensorcnt((short)0);
  }
#else
#pragma unroll
  for (int i = 0; i < 4; ++i) {
    int idx = t + i * 256;
    int row = idx >> 3, cg = (idx & 7) * 8;
    cp_b128(&lQ[row * 72 + cg], &Qb[qbase + (size_t)row * D + cg]);
  }
  wait_cp();
#endif
  __syncthreads();
  const v16bf qf0 = fragA_lds(lQ, 72, w * 16, 0, lane);   // d 0..31
  const v16bf qf1 = fragA_lds(lQ, 72, w * 16, 32, lane);  // d 32..63

  float mrow[8], lrow[8];
  v8f o[4];
#pragma unroll
  for (int j = 0; j < 8; ++j) { mrow[j] = -__builtin_inff(); lrow[j] = 0.f; }
#pragma unroll
  for (int j = 0; j < 4; ++j) o[j] = (v8f){0.f, 0.f, 0.f, 0.f, 0.f, 0.f, 0.f, 0.f};

  for (int k0 = 0; k0 < Kk; k0 += 32) {
    // ---- stage K tile 32x64 (TDM) and V tile transposed 64x32 (manual) ----
#if HAS_TDM
    if (w == 0)
      tdm_load_2d(lds_off(lK), KVb + kvb + (size_t)k0 * 2048, 64, 32, 64, 32,
                  2048, 0x07110000u);
#endif
    {
      int row = t >> 3, cg = (t & 7) * 8;
#if !HAS_TDM
      cp_b128(&lK[row * 72 + cg], &KVb[kvb + (size_t)(k0 + row) * 2048 + cg]);
#endif
      uint4 vv = *(const uint4*)&KVb[kvb + 1024 + (size_t)(k0 + row) * 2048 + cg];
      unsigned uu[4] = {vv.x, vv.y, vv.z, vv.w};
#pragma unroll
      for (int e = 0; e < 4; ++e) {
        lVt[(cg + 2 * e)     * 40 + row] =
            __builtin_bit_cast(__bf16, (unsigned short)(uu[e] & 0xffffu));
        lVt[(cg + 2 * e + 1) * 40 + row] =
            __builtin_bit_cast(__bf16, (unsigned short)(uu[e] >> 16));
      }
    }
#if HAS_TDM
    if (w == 0) __builtin_amdgcn_s_wait_tensorcnt((short)0);
#else
    wait_cp();
#endif
    __syncthreads();

    // ---- S = Q(16x64) @ K^T(64x32): two 16x16 tiles ----
    v8f s0 = (v8f){0.f, 0.f, 0.f, 0.f, 0.f, 0.f, 0.f, 0.f};
    v8f s1 = (v8f){0.f, 0.f, 0.f, 0.f, 0.f, 0.f, 0.f, 0.f};
    s0 = wmma_bf16(qf0, fragB_lds(lK, 72, 0, 0, lane), s0);
    s0 = wmma_bf16(qf1, fragB_lds(lK, 72, 0, 32, lane), s0);
    s1 = wmma_bf16(qf0, fragB_lds(lK, 72, 16, 0, lane), s1);
    s1 = wmma_bf16(qf1, fragB_lds(lK, 72, 16, 32, lane), s1);

    // ---- online softmax (row r of half lives in VGPR r) ----
    float scl[8];
#pragma unroll
    for (int j = 0; j < 8; ++j) {
      float v = fmaxf(s0[j], s1[j]);
      v = fmaxf(v, __shfl_xor(v, 1, 32));
      v = fmaxf(v, __shfl_xor(v, 2, 32));
      v = fmaxf(v, __shfl_xor(v, 4, 32));
      v = fmaxf(v, __shfl_xor(v, 8, 32));
      float mn = fmaxf(mrow[j], v);
      scl[j] = __expf(mrow[j] - mn);
      mrow[j] = mn;
    }
    __bf16* pw = lP[w];
#pragma unroll
    for (int j = 0; j < 8; ++j) {
      float p0 = __expf(s0[j] - mrow[j]);
      float p1 = __expf(s1[j] - mrow[j]);
      pw[(j + 8 * half) * 36 + n]      = (__bf16)p0;
      pw[(j + 8 * half) * 36 + 16 + n] = (__bf16)p1;
      float rs = p0 + p1;
      rs += __shfl_xor(rs, 1, 32);
      rs += __shfl_xor(rs, 2, 32);
      rs += __shfl_xor(rs, 4, 32);
      rs += __shfl_xor(rs, 8, 32);
      lrow[j] = lrow[j] * scl[j] + rs;
    }
#pragma unroll
    for (int nt = 0; nt < 4; ++nt)
#pragma unroll
      for (int j = 0; j < 8; ++j) o[nt][j] *= scl[j];

    // ---- O(16x64) += P(16x32) @ V(32x64) ----
    v16bf pf = fragA_lds(pw, 36, 0, 0, lane);
#pragma unroll
    for (int nt = 0; nt < 4; ++nt)
      o[nt] = wmma_bf16(pf, fragB_lds(lVt, 40, nt * 16, 0, lane), o[nt]);

    __syncthreads();  // protect lK/lVt before next staging
  }

  // ---- normalize and store ----
  float inv[8];
#pragma unroll
  for (int j = 0; j < 8; ++j) inv[j] = 1.0f / lrow[j];
  const size_t obase = ((size_t)b * Nq + nt0 + w * 16) * D + h * 64;
#pragma unroll
  for (int nt = 0; nt < 4; ++nt)
#pragma unroll
    for (int j = 0; j < 8; ++j)
      Ob[obase + (size_t)(j + 8 * half) * D + nt * 16 + n] = (__bf16)(o[nt][j] * inv[j]);
}

// ---------------------------------------------------------------------------
// Host launcher
// ---------------------------------------------------------------------------
extern "C" void kernel_launch(void* const* d_in, const int* in_sizes, int n_in,
                              void* d_out, int out_size, void* d_ws, size_t ws_size,
                              hipStream_t stream) {
  const float* x     = (const float*)d_in[0];
  const float* ctx   = (const float*)d_in[1];
  const float* Wq    = (const float*)d_in[2];
  const float* bq    = (const float*)d_in[3];
  const float* Wkv   = (const float*)d_in[4];
  const float* bkv   = (const float*)d_in[5];
  const float* Wproj = (const float*)d_in[6];
  const float* bproj = (const float*)d_in[7];
  float* out = (float*)d_out;

  const int  Bb = 2, Nq = 2048, Kk = 2048, D = 1024;
  const long nx = (long)Bb * Nq * D;  // 4,194,304

  __bf16* xb     = (__bf16*)d_ws;
  __bf16* ctxb   = xb + nx;
  __bf16* Qb     = ctxb + nx;
  __bf16* Ob     = Qb + nx;
  __bf16* KVb    = Ob + nx;            // 2*nx elems
  __bf16* WqT    = KVb + 2 * nx;
  __bf16* WkvT   = WqT + (long)D * D;
  __bf16* WprojT = WkvT + (long)D * 2 * D;

  cvt_f32_bf16_v4<<<(int)(nx / 1024), 256, 0, stream>>>((const float4*)x,   (ushort4*)xb,   nx / 4);
  cvt_f32_bf16_v4<<<(int)(nx / 1024), 256, 0, stream>>>((const float4*)ctx, (ushort4*)ctxb, nx / 4);

  transpose_cvt<<<dim3(D / 32, D / 32),     256, 0, stream>>>(Wq,    WqT,    D, D);
  transpose_cvt<<<dim3(2 * D / 32, D / 32), 256, 0, stream>>>(Wkv,   WkvT,   D, 2 * D);
  transpose_cvt<<<dim3(D / 32, D / 32),     256, 0, stream>>>(Wproj, WprojT, D, D);

  gemm_bf16<<<dim3(D / 128, (Bb * Nq) / 128), 256, 0, stream>>>(
      xb, WqT, bq, Qb, nullptr, 0.125f, D, D);   // SCALE = 64^-0.5 folded in
  gemm_bf16<<<dim3(2 * D / 128, (Bb * Kk) / 128), 256, 0, stream>>>(
      ctxb, WkvT, bkv, KVb, nullptr, 1.0f, 2 * D, D);

  attn_kernel<<<dim3(Nq / 128, 16, Bb), 256, 0, stream>>>(Qb, KVb, Ob, Nq, Kk);

  gemm_bf16<<<dim3(D / 128, (Bb * Nq) / 128), 256, 0, stream>>>(
      Ob, WprojT, bproj, nullptr, out, 1.0f, D, D);
}
